// VectorQuantize_62440234549776
// MI455X (gfx1250) — matrix operand order repaired and checked by reading, not compiled, not run on hardware
//
#include <hip/hip_runtime.h>

// ---------------------------------------------------------------------------
// VQ-VAE EuclideanCodebook forward + EMA update for MI455X (gfx1250, wave32).
// Distance GEMM (32768x8192x256 = 137 GFLOP, compute-bound) runs on
// v_wmma_f32_16x16x32_bf16 with a fused argmin epilogue. All value-carrying
// outputs (quantize, loss, EMA buffers) stay in exact f32.
// ---------------------------------------------------------------------------

#define DIMD   256       // embedding dim (compile-time for layout)
#define DECAY  0.1f
#define EPS    1e-5f

typedef __attribute__((ext_vector_type(16))) __bf16 v16bf;
typedef __attribute__((ext_vector_type(8)))  float  v8f;

__device__ __forceinline__ unsigned short f32_to_bf16_rne(float f) {
    unsigned int u = __float_as_uint(f);
    unsigned int r = u + 0x7FFFu + ((u >> 16) & 1u);   // round-to-nearest-even
    return (unsigned short)(r >> 16);
}

__device__ __forceinline__ void cvt4(const float4 p, unsigned short* o) {
    o[0] = f32_to_bf16_rne(p.x); o[1] = f32_to_bf16_rne(p.y);
    o[2] = f32_to_bf16_rne(p.z); o[3] = f32_to_bf16_rne(p.w);
}

// --------------------------- workspace zero fill ---------------------------
__global__ void zero_ws_kernel(float* p, long n) {
    long i = (long)blockIdx.x * blockDim.x + threadIdx.x;
    if (i < n) p[i] = 0.0f;
}

// ---------------- kernel A: embed -> bf16 copy + e2[k] = ||e_k||^2 ---------
__global__ __launch_bounds__(256) void prep_embed_kernel(
    const float* __restrict__ embed, unsigned short* __restrict__ eb,
    float* __restrict__ e2)
{
    int k = blockIdx.x;          // one code row per block
    int d = threadIdx.x;         // 256 threads == DIMD
    float v = embed[(long)k * DIMD + d];
    eb[(long)k * DIMD + d] = f32_to_bf16_rne(v);
    float s = v * v;
    #pragma unroll
    for (int o = 16; o > 0; o >>= 1) s += __shfl_xor(s, o, 32);
    __shared__ float red[8];
    if ((d & 31) == 0) red[d >> 5] = s;
    __syncthreads();
    if (d == 0) {
        float t = 0.f;
        #pragma unroll
        for (int i = 0; i < 8; ++i) t += red[i];
        e2[k] = t;
    }
}

// ---------------- kernel B: WMMA distance GEMM + fused argmin --------------
// Block = 256 threads = 8 waves; each wave owns 16 x-rows (block: 128 rows).
// A-fragments (16x32 bf16) live in registers for the whole K sweep; B tiles
// (32x16 bf16, 16 codes) stream from the L2-resident bf16 codebook.
__global__ __launch_bounds__(256) void argmin_gemm_kernel(
    const float* __restrict__ x, const unsigned short* __restrict__ eb,
    const float* __restrict__ e2, int K,
    int* __restrict__ ind, float* __restrict__ ind_f)
{
    const int wave = threadIdx.x >> 5;
    const int lane = threadIdx.x & 31;
    const int col  = lane & 15;        // C/B column within tile
    const int hi   = lane >> 4;        // lane-group (K-range select)
    const int n0   = blockIdx.x * 128 + wave * 16;
    const int row  = n0 + col;         // this lane's A row

    // ---- load 8 A fragments (16x32 bf16, contiguous k order, same as B) ----
    // lane l (row = l%16): elems 0..15 = K d0 + 16*hi + [0..15]
    v16bf A[8];
    {
        const float4* xr = (const float4*)(x + (long)row * DIMD);
        #pragma unroll
        for (int f = 0; f < 8; ++f) {
            const int d0 = f * 32;
            union { unsigned short us[16]; v16bf v; } a;
            float4 p0 = xr[(d0 + hi * 16) >> 2];
            float4 p1 = xr[((d0 + hi * 16) >> 2) + 1];
            float4 p2 = xr[((d0 + hi * 16) >> 2) + 2];
            float4 p3 = xr[((d0 + hi * 16) >> 2) + 3];
            cvt4(p0, &a.us[0]);  cvt4(p1, &a.us[4]);
            cvt4(p2, &a.us[8]);  cvt4(p3, &a.us[12]);
            A[f] = a.v;
        }
    }

    float mind[8];
    int   marg[8];
    #pragma unroll
    for (int v = 0; v < 8; ++v) { mind[v] = 3.4028235e38f; marg[v] = 0; }

    for (int k0 = 0; k0 < K; k0 += 16) {
        const int kcol = k0 + col;
        // prefetch next tile's codebook row for this lane (global_prefetch_b8)
        if (k0 + 16 < K)
            __builtin_prefetch(eb + (long)(kcol + 16) * DIMD, 0, 0);

        v8f c = {};
        const unsigned short* bbase = eb + (long)kcol * DIMD + hi * 16;
        #pragma unroll
        for (int f = 0; f < 8; ++f) {
            // B 32x16 layout: lanes 0-15 K=0..15, lanes 16-31 K=16..31 (32B/lane)
            v16bf B = *(const v16bf*)(bbase + f * 32);
            c = __builtin_amdgcn_wmma_f32_16x16x32_bf16(
                    false, A[f], false, B, (short)0, c, false, false);
        }
        const float ek = e2[kcol];
        #pragma unroll
        for (int v = 0; v < 8; ++v) {
            float dist = ek - 2.0f * c[v];   // ||x||^2 row-constant: dropped
            if (dist < mind[v]) { mind[v] = dist; marg[v] = kcol; }
        }
    }

    // ---- reduce across the 16 lanes of each half-wave (same x-row set) ----
    #pragma unroll
    for (int v = 0; v < 8; ++v) {
        float d = mind[v]; int a = marg[v];
        #pragma unroll
        for (int s = 1; s < 16; s <<= 1) {
            float od = __shfl_xor(d, s, 32);
            int   oa = __shfl_xor(a, s, 32);
            if (od < d || (od == d && oa < a)) { d = od; a = oa; }
        }
        if (col == 0) {               // lanes 0 & 16 write rows v and v+8
            int r = n0 + v + hi * 8;
            ind[r]   = a;
            ind_f[r] = (float)a;
        }
    }
}

// ------- kernel C: gather quantize, commit-loss, bins + embed_sum ----------
__global__ __launch_bounds__(256) void gather_scatter_kernel(
    const float* __restrict__ x, const float* __restrict__ embed,
    const int* __restrict__ ind, float* __restrict__ out_q,
    float* loss_acc, float* __restrict__ bins, float* __restrict__ embed_sum)
{
    int n = blockIdx.x;
    int d = threadIdx.x;
    int k = ind[n];
    float q  = embed[(long)k * DIMD + d];
    float xv = x[(long)n * DIMD + d];
    out_q[(long)n * DIMD + d] = q;          // quantize_st == quantize fwd value
    float df = q - xv;
    float s = df * df;
    #pragma unroll
    for (int o = 16; o > 0; o >>= 1) s += __shfl_xor(s, o, 32);
    __shared__ float red[8];
    if ((d & 31) == 0) red[d >> 5] = s;
    __syncthreads();
    if (d == 0) {
        float t = 0.f;
        #pragma unroll
        for (int i = 0; i < 8; ++i) t += red[i];
        atomicAdd(loss_acc, t);
        atomicAdd(&bins[k], 1.0f);
    }
    atomicAdd(&embed_sum[(long)k * DIMD + d], xv);   // global_atomic_add_f32
}

// ------- kernel D: new_cluster_size + total reduction ----------------------
__global__ __launch_bounds__(256) void cluster_update_kernel(
    const float* __restrict__ cs, const float* __restrict__ bins,
    float* __restrict__ out_ncs, float* total_acc)
{
    int k = blockIdx.x * 256 + threadIdx.x;
    float ncs = cs[k] * DECAY + bins[k] * (1.0f - DECAY);
    out_ncs[k] = ncs;
    float s = ncs;
    #pragma unroll
    for (int o = 16; o > 0; o >>= 1) s += __shfl_xor(s, o, 32);
    __shared__ float red[8];
    if ((threadIdx.x & 31) == 0) red[threadIdx.x >> 5] = s;
    __syncthreads();
    if (threadIdx.x == 0) {
        float t = 0.f;
        #pragma unroll
        for (int i = 0; i < 8; ++i) t += red[i];
        atomicAdd(total_acc, t);
    }
}

// ------- kernel E: new_embed_avg + laplace-smoothed new_embed --------------
__global__ __launch_bounds__(256) void embed_update_kernel(
    const float* __restrict__ ea, const float* __restrict__ esum,
    const float* __restrict__ ncs, const float* total_acc, int K,
    float* __restrict__ out_nea, float* __restrict__ out_ne)
{
    long i = (long)blockIdx.x * 256 + threadIdx.x;
    int  k = (int)(i >> 8);                     // i / DIMD
    float nea = ea[i] * DECAY + esum[i] * (1.0f - DECAY);
    out_nea[i] = nea;
    float total = *total_acc;
    float smoothed = (ncs[k] + EPS) / (total + (float)K * EPS) * total;
    out_ne[i] = nea / smoothed;
}

// ------- kernel F: finalize commit loss ------------------------------------
__global__ void loss_fin_kernel(const float* loss_acc, float* out_loss, float inv_nd) {
    if (threadIdx.x == 0) *out_loss = (*loss_acc) * inv_nd;
}

// ---------------------------------------------------------------------------
extern "C" void kernel_launch(void* const* d_in, const int* in_sizes, int n_in,
                              void* d_out, int out_size, void* d_ws, size_t ws_size,
                              hipStream_t stream) {
    const float* x    = (const float*)d_in[0];   // [N, D]
    const float* emb  = (const float*)d_in[1];   // [K, D]
    const float* cs   = (const float*)d_in[2];   // [K]
    const float* eavg = (const float*)d_in[3];   // [K, D]

    const int K = in_sizes[2];
    const int N = in_sizes[0] / DIMD;
    const long KD = (long)K * DIMD;
    const long ND = (long)N * DIMD;

    // ---- workspace layout (floats) ----
    float* ws        = (float*)d_ws;
    float* loss_acc  = ws;                       // 1
    float* total_acc = ws + 1;                   // 1
    float* bins      = ws + 2;                   // K
    float* embed_sum = ws + 2 + K;               // K*D
    float* e2        = ws + 2 + K + KD;          // K
    int*   ind       = (int*)(e2 + K);           // N
    unsigned short* eb = (unsigned short*)(ind + N);  // K*D bf16

    // ---- output layout (return-order concat, flat f32) ----
    float* out       = (float*)d_out;
    float* out_q     = out;                      // N*D quantize_st
    float* out_indf  = out + ND;                 // N   embed_ind (as float)
    float* out_loss  = out_indf + N;             // 1   commit_loss
    float* out_ncs   = out_loss + 1;             // K   new_cluster_size
    float* out_nea   = out_ncs + K;              // K*D new_embed_avg
    float* out_ne    = out_nea + KD;             // K*D new_embed

    // 1) zero accumulators (loss, total, bins, embed_sum)
    long nz = 2 + K + KD;
    zero_ws_kernel<<<(int)((nz + 255) / 256), 256, 0, stream>>>(ws, nz);

    // 2) codebook -> bf16 + squared norms
    prep_embed_kernel<<<K, 256, 0, stream>>>(emb, eb, e2);

    // 3) compute-bound distance GEMM on v_wmma_f32_16x16x32_bf16 + argmin
    argmin_gemm_kernel<<<N / 128, 256, 0, stream>>>(x, eb, e2, K, ind, out_indf);

    // 4) gather quantize, commit loss, EMA scatter stats
    gather_scatter_kernel<<<N, 256, 0, stream>>>(x, emb, ind, out_q,
                                                 loss_acc, bins, embed_sum);

    // 5) cluster-size EMA + total mass
    cluster_update_kernel<<<K / 256, 256, 0, stream>>>(cs, bins, out_ncs, total_acc);

    // 6) embed_avg EMA + laplace-smoothed codebook
    embed_update_kernel<<<(int)(KD / 256), 256, 0, stream>>>(eavg, embed_sum,
                                                             out_ncs, total_acc, K,
                                                             out_nea, out_ne);

    // 7) finalize loss
    loss_fin_kernel<<<1, 32, 0, stream>>>(loss_acc, out_loss, 1.0f / (float)ND);
}
